// CrossAttention_33122787787102
// MI455X (gfx1250) — matrix operand order repaired and checked by reading, not compiled, not run on hardware
//
#include <hip/hip_runtime.h>
#include <hip/hip_bf16.h>

typedef __attribute__((ext_vector_type(16))) __bf16 bf16x16;
typedef __attribute__((ext_vector_type(8)))  __bf16 bf16x8;
typedef __attribute__((ext_vector_type(8)))  float  f32x8;

#define HEADS  8
#define DH     64
#define CIN    256
#define HID    512
#define COUT   256
#define LTOK   77
#define DCON   768
#define BATCH  2
#define FRAMES 16
#define HW     4096          // 64*64
#define TILE_N 64
#define NCT    (TILE_N / 16) // 4 col tiles
#define XPAD   8
#define SPAD   8

// ---------------- Stage A: tiny setup kernels ----------------

// kvp[b,l,o] = sum_c kv[b,l,c] * Wkv[o,c]   (o in [0,1024))
__global__ __launch_bounds__(256) void kvproj_kernel(
    const float* __restrict__ kv, const float* __restrict__ Wkv,
    float* __restrict__ kvp) {
  __shared__ float row[DCON];
  const int bl = blockIdx.x;                       // 0 .. B*L-1
  const float* src = kv + (size_t)bl * DCON;
  for (int i = threadIdx.x; i < DCON; i += 256) row[i] = src[i];
  __syncthreads();
#pragma unroll
  for (int j = 0; j < 4; ++j) {
    const int o = threadIdx.x + j * 256;
    const float* wrow = Wkv + (size_t)o * DCON;
    float acc = 0.f;
    for (int c = 0; c < DCON; ++c) acc += row[c] * wrow[c];
    kvp[(size_t)bl * (2 * HID) + o] = acc;
  }
}

// per (b,h): softmax_l(k) then context[d][e] = sum_l ksm[d][l]*v[e][l]
__global__ __launch_bounds__(64) void context_kernel(
    const float* __restrict__ kvp, float* __restrict__ ctx) {
  __shared__ float ksm[DH][LTOK + 3];
  __shared__ float vv [DH][LTOK + 3];
  const int b = blockIdx.x / HEADS, h = blockIdx.x % HEADS;
  const int d = threadIdx.x;
  float m = -1e30f;
  for (int l = 0; l < LTOK; ++l) {
    const size_t base = (size_t)(b * LTOK + l) * (2 * HID);
    float kd = kvp[base + h * DH + d];
    ksm[d][l] = kd;
    vv [d][l] = kvp[base + HID + h * DH + d];
    m = fmaxf(m, kd);
  }
  float s = 0.f;
  for (int l = 0; l < LTOK; ++l) { float e = __expf(ksm[d][l] - m); ksm[d][l] = e; s += e; }
  const float inv = 1.f / s;
  for (int l = 0; l < LTOK; ++l) ksm[d][l] *= inv;
  __syncthreads();
  float* outp = ctx + ((size_t)(b * HEADS + h) * DH + d) * DH;
  for (int e = 0; e < DH; ++e) {
    float acc = 0.f;
    for (int l = 0; l < LTOK; ++l) acc += ksm[d][l] * vv[e][l];
    outp[e] = acc;
  }
}

// M[b,co,h*64+d] = sum_e Wout[co, h*64+e] * ctx[b,h,d,e]   (bf16 out)
__global__ __launch_bounds__(512) void foldm_kernel(
    const float* __restrict__ Wout, const float* __restrict__ ctx,
    __bf16* __restrict__ Mb) {
  __shared__ float wrow[HID];
  const int b = blockIdx.x / COUT, co = blockIdx.x % COUT;
  for (int i = threadIdx.x; i < HID; i += 512) wrow[i] = Wout[(size_t)co * HID + i];
  __syncthreads();
  const int h = threadIdx.x / DH, d = threadIdx.x % DH;
  const float* c = ctx + ((size_t)(b * HEADS + h) * DH + d) * DH;
  float acc = 0.f;
#pragma unroll 8
  for (int e = 0; e < DH; ++e) acc += wrow[h * DH + e] * c[e];
  Mb[(size_t)(b * COUT + co) * HID + threadIdx.x] = (__bf16)acc;
}

__global__ __launch_bounds__(256) void cvt_kernel(
    const float* __restrict__ src, __bf16* __restrict__ dst, int n) {
  int i = blockIdx.x * 256 + threadIdx.x;
  if (i < n) dst[i] = (__bf16)src[i];
}

// ---------------- Stage B: fused GEMM-softmax-GEMM ----------------

// A fragment (16x32 bf16, row-major src, row stride ld):
// lane L: M=L%16; elems 0..7 -> K = kb + 8*(L/16) + e ; elems 8..15 -> +16
__device__ __forceinline__ bf16x16 load_a_frag(const __bf16* base, int row0,
                                               int ld, int kb, int lane) {
  const __bf16* p = base + (size_t)(row0 + (lane & 15)) * ld + kb + ((lane >> 4) << 3);
  bf16x8 lo = *(const bf16x8*)p;
  bf16x8 hi = *(const bf16x8*)(p + 16);
  return __builtin_shufflevector(lo, hi, 0,1,2,3,4,5,6,7,8,9,10,11,12,13,14,15);
}

// B fragment (32x16 bf16) from K-contiguous layout [n][K] with row stride ld:
// lane L: N=L%16; elems 0..7 -> K = kb + 8*(L/16) + e ; elems 8..15 -> +16
__device__ __forceinline__ bf16x16 load_b_frag(const __bf16* base, int n0,
                                               int ld, int kb, int lane) {
  const __bf16* p = base + (size_t)(n0 + (lane & 15)) * ld + kb + ((lane >> 4) << 3);
  bf16x8 lo = *(const bf16x8*)p;
  bf16x8 hi = *(const bf16x8*)(p + 16);
  return __builtin_shufflevector(lo, hi, 0,1,2,3,4,5,6,7,8,9,10,11,12,13,14,15);
}

__global__ __launch_bounds__(256) void fused_attn_kernel(
    const float* __restrict__ x, const __bf16* __restrict__ Wqb,
    const __bf16* __restrict__ Mb, const float* __restrict__ bout,
    float* __restrict__ out) {
  __shared__ __bf16 XT[TILE_N][CIN + XPAD];          // x tile, transposed, bf16 (33,792 B)
  __shared__ union SBu {
    __bf16 st[TILE_N][HID + SPAD];                   // softmaxed S, transposed (66,560 B)
    float  ob[COUT][TILE_N];                         // staged output tile     (65,536 B)
  } SB;

  const int tilesPerImg = HW / TILE_N;
  const int img  = blockIdx.x / tilesPerImg;
  const int pix0 = (blockIdx.x % tilesPerImg) * TILE_N;
  const int b = img / FRAMES, f = img % FRAMES;

  const float* xbase = x + (size_t)b * CIN * FRAMES * HW + (size_t)f * HW + pix0;

  // Phase 1: stage bf16 X tile transposed into LDS (coalesced global reads)
  for (int i = threadIdx.x; i < CIN * TILE_N; i += 256) {
    const int c = i / TILE_N, p = i % TILE_N;
    XT[p][c] = (__bf16)xbase[(size_t)c * (FRAMES * HW) + p];
  }
  __syncthreads();

  const int wave = threadIdx.x >> 5;   // == head
  const int lane = threadIdx.x & 31;

  // ---- GEMM1: S[head rows 64 x TILE_N pix] = Wq @ X ----
  f32x8 acc[4][NCT] = {};
  const int row0 = wave * DH;
#pragma unroll
  for (int k = 0; k < CIN / 32; ++k) {
    const int kb = k * 32;
    bf16x16 afr[4];
#pragma unroll
    for (int rt = 0; rt < 4; ++rt) afr[rt] = load_a_frag(Wqb, row0 + rt * 16, CIN, kb, lane);
#pragma unroll
    for (int ct = 0; ct < NCT; ++ct) {
      bf16x16 bfr = load_b_frag(&XT[0][0], ct * 16, CIN + XPAD, kb, lane);
#pragma unroll
      for (int rt = 0; rt < 4; ++rt)
        acc[rt][ct] = __builtin_amdgcn_wmma_f32_16x16x32_bf16(
            false, afr[rt], false, bfr, (short)0, acc[rt][ct], false, false);
    }
  }

  // ---- in-register softmax over the head's 64 rows, per pixel column ----
  // C layout: VGPR r, lane L -> M = r + 8*(L/16) (+16*rt), N = L%16.
  // Column N lives entirely in lanes {N, N+16}: local reduce + one shfl_xor(16).
#pragma unroll
  for (int ct = 0; ct < NCT; ++ct) {
    float m = -1e30f;
#pragma unroll
    for (int rt = 0; rt < 4; ++rt)
#pragma unroll
      for (int r = 0; r < 8; ++r) m = fmaxf(m, acc[rt][ct][r]);
    m = fmaxf(m, __shfl_xor(m, 16, 32));
    float s = 0.f;
#pragma unroll
    for (int rt = 0; rt < 4; ++rt)
#pragma unroll
      for (int r = 0; r < 8; ++r) {
        float e = __expf(acc[rt][ct][r] - m);
        acc[rt][ct][r] = e;
        s += e;
      }
    s += __shfl_xor(s, 16, 32);
    const float inv = 0.125f / s;          // * dh^-0.5
#pragma unroll
    for (int rt = 0; rt < 4; ++rt) {
      bf16x8 v;
#pragma unroll
      for (int r = 0; r < 8; ++r) v[r] = (__bf16)(acc[rt][ct][r] * inv);
      const int n  = ct * 16 + (lane & 15);
      const int kk = wave * DH + rt * 16 + ((lane >> 4) << 3);
      *(bf16x8*)&SB.st[n][kk] = v;         // 8 consecutive K -> one 16B ds_store
    }
  }
  __syncthreads();

  // ---- GEMM2: O[256 x TILE_N] = M[b] @ S ----
  f32x8 acc2[2][NCT] = {};
  const __bf16* mrow = Mb + (size_t)b * COUT * HID;
  const int row2 = wave * 32;
#pragma unroll
  for (int k = 0; k < HID / 32; ++k) {
    const int kb = k * 32;
    bf16x16 afr[2];
#pragma unroll
    for (int j = 0; j < 2; ++j) afr[j] = load_a_frag(mrow, row2 + j * 16, HID, kb, lane);
#pragma unroll
    for (int ct = 0; ct < NCT; ++ct) {
      bf16x16 bfr = load_b_frag(&SB.st[0][0], ct * 16, HID + SPAD, kb, lane);
#pragma unroll
      for (int j = 0; j < 2; ++j)
        acc2[j][ct] = __builtin_amdgcn_wmma_f32_16x16x32_bf16(
            false, afr[j], false, bfr, (short)0, acc2[j][ct], false, false);
    }
  }
  __syncthreads();   // all waves done reading SB.st -> safe to overlay O tile

  // stage O tile in LDS for coalesced writeout
#pragma unroll
  for (int j = 0; j < 2; ++j)
#pragma unroll
    for (int ct = 0; ct < NCT; ++ct)
#pragma unroll
      for (int r = 0; r < 8; ++r) {
        const int co = row2 + j * 16 + r + ((lane >> 4) << 3);
        const int n  = ct * 16 + (lane & 15);
        SB.ob[co][n] = acc2[j][ct][r];
      }
  __syncthreads();

  // coalesced b128 stores: 16 consecutive threads cover one 256B output row
  float* obase = out + (size_t)b * COUT * FRAMES * HW + (size_t)f * HW + pix0;
  const int CHUNKS = COUT * TILE_N / 4 / 256;   // 16 float4 chunks per thread
#pragma unroll
  for (int i = 0; i < CHUNKS; ++i) {
    const int it = i * 256 + threadIdx.x;
    const int co = it >> 4;                     // TILE_N/4 = 16 chunks per row
    const int nq = (it & 15) << 2;
    float4 v = *(const float4*)&SB.ob[co][nq];
    const float bb = bout[co];
    v.x += bb; v.y += bb; v.z += bb; v.w += bb;
    *(float4*)(obase + (size_t)co * (FRAMES * HW) + nq) = v;
  }
}

// ---------------- launch ----------------
extern "C" void kernel_launch(void* const* d_in, const int* in_sizes, int n_in,
                              void* d_out, int out_size, void* d_ws, size_t ws_size,
                              hipStream_t stream) {
  (void)in_sizes; (void)n_in; (void)out_size; (void)ws_size;
  const float* x    = (const float*)d_in[0];
  const float* kv   = (const float*)d_in[1];
  const float* Wq   = (const float*)d_in[2];
  const float* Wkv  = (const float*)d_in[3];
  const float* Wout = (const float*)d_in[4];
  const float* bout = (const float*)d_in[5];
  float* out = (float*)d_out;

  // workspace layout (all offsets 256B-aligned), total ~1.68 MB
  char* ws = (char*)d_ws;
  float*  kvp = (float*)(ws);                                  // 2*77*1024 f32  = 630,784 B
  float*  ctx = (float*)(ws + 630784);                         // 2*8*64*64 f32  = 262,144 B
  __bf16* Mb  = (__bf16*)(ws + 630784 + 262144);               // 2*256*512 bf16 = 524,288 B
  __bf16* Wqb = (__bf16*)(ws + 630784 + 262144 + 524288);      // 512*256 bf16   = 262,144 B

  kvproj_kernel <<<BATCH * LTOK,  256, 0, stream>>>(kv, Wkv, kvp);
  context_kernel<<<BATCH * HEADS,  64, 0, stream>>>(kvp, ctx);
  foldm_kernel  <<<BATCH * COUT,  512, 0, stream>>>(Wout, ctx, Mb);
  cvt_kernel    <<<(HID * CIN + 255) / 256, 256, 0, stream>>>(Wq, Wqb, HID * CIN);
  fused_attn_kernel<<<BATCH * FRAMES * (HW / TILE_N), 256, 0, stream>>>(x, Wqb, Mb, bout, out);
}
